// SlotClassifier_19945828122750
// MI455X (gfx1250) — matrix operand order repaired and checked by reading, not compiled, not run on hardware
//
#include <hip/hip_runtime.h>
#include <hip/hip_bf16.h>
#include <math.h>

// ---------------------------------------------------------------------------
// CDNA5 (gfx1250) wave32 WMMA types
// ---------------------------------------------------------------------------
typedef __bf16 bf16_t;
typedef __attribute__((ext_vector_type(16))) __bf16 v16bf;
typedef __attribute__((ext_vector_type(8)))  float  v8f;
typedef __attribute__((ext_vector_type(4)))  unsigned u32x4;

union Frag { v16bf v; u32x4 q[2]; };

__device__ __forceinline__ bf16_t bfzero() {
    union { unsigned short s; bf16_t b; } u; u.s = 0; return u.b;
}

// float -> bf16 round-to-nearest-even
__device__ __forceinline__ bf16_t f2bf(float f) {
    union { float f; unsigned u; } v; v.f = f;
    unsigned r = v.u + 0x7FFFu + ((v.u >> 16) & 1u);
    union { unsigned short s; bf16_t b; } o; o.s = (unsigned short)(r >> 16);
    return o.b;
}

__device__ __forceinline__ float gelu_exact(float x) {
    return 0.5f * x * (1.0f + erff(x * 0.70710678118654752f));
}

// 16-byte bulk copy helper (global <-> LDS), count in u32x4 units
__device__ __forceinline__ void copy_v4(u32x4* __restrict__ dst, const u32x4* __restrict__ src,
                                        int nvec, int tid, int nth) {
    for (int i = tid; i < nvec; i += nth) dst[i] = src[i];
}

// ---------------------------------------------------------------------------
// Tiled GEMM: C = A(Mpad x K, bf16 LDS, row-major, rows padded/zeroed to Mt*16)
//           * Bpk(K x N bf16 global, k-block-packed: [(k/32)*N + n][k%32])
// K % 32 == 0, N % 16 == 0. Per k-step: 2x ds_load_b128 + 2x global_load_b128
// + 1x v_wmma_f32_16x16x32_bf16. Epilogue gets every (row, col, acc);
// epilogues writing global/RES guard rows themselves, LDS writers don't.
// ---------------------------------------------------------------------------
template <typename EPI>
__device__ __forceinline__ void gemm_tiles(const bf16_t* __restrict__ A, int lda, int Mt,
                                           const bf16_t* __restrict__ Bpk, int N, int K, int Nt,
                                           EPI epi) {
    const int wave   = threadIdx.x >> 5;
    const int lane   = threadIdx.x & 31;
    const int nwaves = blockDim.x >> 5;
    const int ml = lane & 15;
    const int hi = lane >> 4;

    for (int t = wave; t < Mt * Nt; t += nwaves) {
        const int m0 = (t / Nt) << 4;
        const int n0 = (t % Nt) << 4;
        const bf16_t* arow = A + (m0 + ml) * lda + 8 * hi;
        const bf16_t* bcol = Bpk + ((n0 + ml) << 5) + 8 * hi;
        __builtin_prefetch(bcol, 0, 1);
        v8f c = {};
        for (int k0 = 0; k0 < K; k0 += 32) {
            Frag a, b;
            a.q[0] = *(const u32x4*)(arow + k0);
            a.q[1] = *(const u32x4*)(arow + k0 + 16);
            const bf16_t* bp = bcol + ((k0 >> 5) * N << 5);
            b.q[0] = *(const u32x4*)(bp);
            b.q[1] = *(const u32x4*)(bp + 16);
            c = __builtin_amdgcn_wmma_f32_16x16x32_bf16(false, a.v, false, b.v,
                                                        (short)0, c, false, false);
        }
#pragma unroll
        for (int v = 0; v < 8; ++v) epi(m0 + v + 8 * hi, n0 + ml, c[v]);
    }
}

// ---------------------------------------------------------------------------
// Workspace layout (bytes)
// ---------------------------------------------------------------------------
#define WS_CONVB   0            // bf16 packed [96 ][160]
#define WS_PROJB   30720        // bf16 packed [160][64 ]
#define WS_QKVB    51200        // bf16 packed [64 ][192]
#define WS_OUTB    75776        // bf16 packed [64 ][64 ]
#define WS_FF1B    83968        // bf16 packed [64 ][256]
#define WS_FF2B    116736       // bf16 packed [256][64 ]
#define WS_TOKENS  149504       // f32  [512][196][64]

// ---------------------------------------------------------------------------
// Kernel P: repack all weights into bf16 k-block-packed B layouts (zero pad)
// packed index: ((k>>5)*N + n)*32 + (k&31)
// ---------------------------------------------------------------------------
__device__ __forceinline__ void pk_store(bf16_t* dst, int N, int k, int n, float val) {
    dst[(((k >> 5) * N + n) << 5) + (k & 31)] = f2bf(val);
}

__global__ void sc_prep(const float* __restrict__ conv_w, const float* __restrict__ proj_w,
                        const float* __restrict__ qkv_w,  const float* __restrict__ out_w,
                        const float* __restrict__ ff1_w,  const float* __restrict__ ff2_w,
                        bf16_t* convB, bf16_t* projB, bf16_t* qkvB,
                        bf16_t* outB,  bf16_t* ff1B,  bf16_t* ff2B) {
    int idx = blockIdx.x * blockDim.x + threadIdx.x;
    if (idx < 96 * 160) { int k = idx / 160, n = idx % 160;
        pk_store(convB, 160, k, n, (k < 81 && n < 150) ? conv_w[n * 81 + k] : 0.f); return; }
    idx -= 96 * 160;
    if (idx < 160 * 64) { int k = idx / 64, n = idx % 64;
        pk_store(projB, 64, k, n, (k < 150) ? proj_w[n * 150 + k] : 0.f); return; }
    idx -= 160 * 64;
    if (idx < 64 * 192) { int k = idx / 192, n = idx % 192;
        pk_store(qkvB, 192, k, n, qkv_w[n * 64 + k]); return; }
    idx -= 64 * 192;
    if (idx < 64 * 64) { int k = idx / 64, n = idx % 64;
        pk_store(outB, 64, k, n, out_w[n * 64 + k]); return; }
    idx -= 64 * 64;
    if (idx < 64 * 256) { int k = idx / 256, n = idx % 256;
        pk_store(ff1B, 256, k, n, ff1_w[n * 64 + k]); return; }
    idx -= 64 * 256;
    if (idx < 256 * 64) { int k = idx / 64, n = idx % 64;
        pk_store(ff2B, 64, k, n, ff2_w[n * 256 + k]); return; }
}

// ---------------------------------------------------------------------------
// Kernel 1: conv(9x9,s2,p4)+GELU as im2col GEMM, then 1x1 proj GEMM.
// One block per image. All LDS tiles padded to 208 rows (zero rows => no
// guards in fragment loads).
// ---------------------------------------------------------------------------
#define K1_IMG   0
#define K1_PATCH 3136                    // 208 x 96 bf16  (39936)
#define K1_HBUF  43072                   // 208 x 160 bf16 (66560)
#define K1_LDS   109632

__global__ void sc_conv_proj(const float* __restrict__ x,
                             const bf16_t* __restrict__ convB,
                             const bf16_t* __restrict__ projB,
                             const float* __restrict__ proj_b,
                             float* __restrict__ tokens) {
    extern __shared__ char smem[];
    float*  IMG   = (float*)(smem + K1_IMG);
    bf16_t* PATCH = (bf16_t*)(smem + K1_PATCH);
    bf16_t* HBUF  = (bf16_t*)(smem + K1_HBUF);
    const int b = blockIdx.x, tid = threadIdx.x, nth = blockDim.x;

    copy_v4((u32x4*)IMG, (const u32x4*)(x + (size_t)b * 784), 784 / 4, tid, nth);
    __syncthreads();

    // im2col, zero padded (k: 81->96, rows: 196->208)
    for (int i = tid; i < 208 * 96; i += nth) {
        const int p = i / 96, k = i % 96;
        float v = 0.f;
        if (p < 196 && k < 81) {
            const int ky = k / 9, kx = k % 9;
            const int oy = p / 14, ox = p % 14;
            const int iy = oy * 2 - 4 + ky, ix = ox * 2 - 4 + kx;
            if (iy >= 0 && iy < 28 && ix >= 0 && ix < 28) v = IMG[iy * 28 + ix];
        }
        PATCH[i] = f2bf(v);
    }
    __syncthreads();

    // conv GEMM: (208x96)*(96x160) -> GELU -> HBUF (pad rows compute to 0)
    gemm_tiles(PATCH, 96, 13, convB, 160, 96, 10,
               [&](int r, int c, float v) { HBUF[r * 160 + c] = f2bf(gelu_exact(v)); });
    __syncthreads();

    // proj GEMM: (208x160)*(160x64) + bias -> tokens f32 (global, guarded)
    float* tout = tokens + (size_t)b * 196 * 64;
    gemm_tiles(HBUF, 160, 13, projB, 64, 160, 4,
               [&](int r, int c, float v) { if (r < 196) tout[r * 64 + c] = v + proj_b[c]; });
}

// ---------------------------------------------------------------------------
// Kernel 2: pre-norm transformer layer, one block (8 waves) per image.
// LDS: RES f32[196x64] | ABUF bf16[208x64] | Q bf16[208x64] | K bf16[208x64]
//    | VT bf16[64x224] | BIG (attn scratch 8x(16x208 f32 + 16x224 bf16) or
//      FFN hidden bf16[208x256])
// ---------------------------------------------------------------------------
#define K2_RES   0
#define K2_ABUF  50176
#define K2_Q     76800
#define K2_K     103424
#define K2_VT    130048
#define K2_BIG   158720
#define K2_LDS   (158720 + 163840)

__device__ __forceinline__ void ln_rows_bf16(const float* __restrict__ src, bf16_t* __restrict__ dst,
                                             const float* __restrict__ s, const float* __restrict__ bsh,
                                             int tid, int nth) {
    for (int r = tid; r < 196; r += nth) {
        const float* xr = src + r * 64;
        float mu = 0.f;
        for (int i = 0; i < 64; ++i) mu += xr[i];
        mu *= (1.f / 64.f);
        float var = 0.f;
        for (int i = 0; i < 64; ++i) { float d = xr[i] - mu; var += d * d; }
        var *= (1.f / 64.f);
        const float rs = rsqrtf(var + 1e-5f);
        for (int i = 0; i < 64; ++i) dst[r * 64 + i] = f2bf((xr[i] - mu) * rs * s[i] + bsh[i]);
    }
}

__global__ void sc_transformer(float* __restrict__ tokens,
                               const bf16_t* __restrict__ qkvB, const bf16_t* __restrict__ outB,
                               const bf16_t* __restrict__ ff1B, const bf16_t* __restrict__ ff2B,
                               const float* __restrict__ ln1_s, const float* __restrict__ ln1_b,
                               const float* __restrict__ attn_in_b, const float* __restrict__ attn_out_b,
                               const float* __restrict__ ln2_s, const float* __restrict__ ln2_b,
                               const float* __restrict__ ffn_b1, const float* __restrict__ ffn_b2,
                               const float* __restrict__ tln_s, const float* __restrict__ tln_b) {
    extern __shared__ char smem[];
    float*  RES  = (float*)(smem + K2_RES);
    bf16_t* ABUF = (bf16_t*)(smem + K2_ABUF);
    bf16_t* Qb   = (bf16_t*)(smem + K2_Q);
    bf16_t* Kb   = (bf16_t*)(smem + K2_K);
    bf16_t* VT   = (bf16_t*)(smem + K2_VT);
    char*   BIG  = smem + K2_BIG;

    const int b = blockIdx.x, tid = threadIdx.x, nth = blockDim.x;
    float* tg = tokens + (size_t)b * 196 * 64;

    copy_v4((u32x4*)RES, (const u32x4*)tg, 196 * 64 / 4, tid, nth);
    // zero pads: ABUF rows 196..207 (A operand of QKV GEMM), VT cols 208..223
    for (int i = tid; i < 12 * 64; i += nth) ABUF[196 * 64 + i] = bfzero();
    for (int i = tid; i < 64 * 16; i += nth) VT[(i >> 4) * 224 + 208 + (i & 15)] = bfzero();
    __syncthreads();

    // ---- LN1 -> ABUF ----
    ln_rows_bf16(RES, ABUF, ln1_s, ln1_b, tid, nth);
    __syncthreads();

    // ---- QKV GEMM: (208x64)*(64x192)+bias -> Q[208x64], K[208x64], VT[64x224]
    gemm_tiles(ABUF, 64, 13, qkvB, 192, 64, 12,
               [&](int r, int c, float v) {
                   const float val = v + attn_in_b[c];
                   if (c < 64)       Qb[r * 64 + c]          = f2bf(val);
                   else if (c < 128) Kb[r * 64 + (c - 64)]   = f2bf(val);
                   else              VT[(c - 128) * 224 + r] = f2bf(val);
               });
    __syncthreads();

    // ---- attention: wave owns (head, 16-row m-tile); dh=16 zero-padded to K=32
    {
        const int wave = tid >> 5, lane = tid & 31;
        const int ml = lane & 15, hi = lane >> 4;
        float*  srow = (float*)(BIG + wave * 20480);           // 16 x 208 f32
        bf16_t* pbuf = (bf16_t*)(BIG + wave * 20480 + 13312);  // 16 x 224 bf16
        const u32x4 zq = {0u, 0u, 0u, 0u};
        for (int item = wave; item < 52; item += 8) {
            const int h = item & 3, mt = item >> 2;
            const int m0 = mt * 16;
            // Q fragment: k = 8*hi..8*hi+7 valid (dh=16), upper K half zero
            Frag a;
            a.q[0] = *(const u32x4*)(Qb + (m0 + ml) * 64 + 16 * h + 8 * hi);
            a.q[1] = zq;
            for (int nt = 0; nt < 13; ++nt) {
                const int n0 = nt * 16;
                Frag bv;                    // K^T fragment: k = 8*hi..8*hi+7 valid, upper K half zero
                const bf16_t* kr = Kb + (n0 + ml) * 64 + 16 * h;
                bv.q[0] = *(const u32x4*)(kr + 8 * hi);
                bv.q[1] = zq;
                v8f c = {};
                c = __builtin_amdgcn_wmma_f32_16x16x32_bf16(false, a.v, false, bv.v,
                                                            (short)0, c, false, false);
#pragma unroll
                for (int v = 0; v < 8; ++v) srow[(v + 8 * hi) * 208 + n0 + ml] = c[v];
            }
            // in-wave row softmax (scale 0.25 = 1/sqrt(dh)); bf16 probs, pads = 0
            if (lane < 16) {
                float* r = srow + lane * 208;
                float mx = -1e30f;
                for (int j = 0; j < 196; ++j) mx = fmaxf(mx, r[j] * 0.25f);
                float sm = 0.f;
                for (int j = 0; j < 196; ++j) { float e = expf(r[j] * 0.25f - mx); r[j] = e; sm += e; }
                const float inv = 1.f / fmaxf(sm, 1e-20f);
                bf16_t* pr = pbuf + lane * 224;
                for (int j = 0; j < 224; ++j) pr[j] = (j < 196) ? f2bf(r[j] * inv) : bfzero();
            }
            // O = P @ V  (K = 224 padded; VT is [dim][token] so B-frags contiguous)
            v8f c = {};
            for (int k0 = 0; k0 < 224; k0 += 32) {
                Frag a2, bv;
                const bf16_t* pr = pbuf + ml * 224 + k0 + 8 * hi;
                a2.q[0] = *(const u32x4*)(pr);
                a2.q[1] = *(const u32x4*)(pr + 16);
                const bf16_t* vr = VT + (16 * h + ml) * 224 + k0 + 8 * hi;
                bv.q[0] = *(const u32x4*)(vr);
                bv.q[1] = *(const u32x4*)(vr + 16);
                c = __builtin_amdgcn_wmma_f32_16x16x32_bf16(false, a2.v, false, bv.v,
                                                            (short)0, c, false, false);
            }
#pragma unroll
            for (int v = 0; v < 8; ++v)
                ABUF[(m0 + v + 8 * hi) * 64 + 16 * h + ml] = f2bf(c[v]);  // padded rows ok
        }
    }
    __syncthreads();

    // ---- out proj + residual ----
    gemm_tiles(ABUF, 64, 13, outB, 64, 64, 4,
               [&](int r, int c, float v) { if (r < 196) RES[r * 64 + c] += v + attn_out_b[c]; });
    __syncthreads();

    // ---- LN2 -> ABUF ----
    ln_rows_bf16(RES, ABUF, ln2_s, ln2_b, tid, nth);
    __syncthreads();

    // ---- FFN1 + GELU -> FF1 (BIG reused; all 208 rows written, finite) ----
    bf16_t* FF1 = (bf16_t*)BIG;   // 208 x 256
    gemm_tiles(ABUF, 64, 13, ff1B, 256, 64, 16,
               [&](int r, int c, float v) { FF1[r * 256 + c] = f2bf(gelu_exact(v + ffn_b1[c])); });
    __syncthreads();

    // ---- FFN2 + residual ----
    gemm_tiles(FF1, 256, 13, ff2B, 64, 256, 4,
               [&](int r, int c, float v) { if (r < 196) RES[r * 64 + c] += v + ffn_b2[c]; });
    __syncthreads();

    // ---- final LN (tln) -> tokens (global f32) ----
    for (int r = tid; r < 196; r += nth) {
        const float* xr = RES + r * 64;
        float mu = 0.f;
        for (int i = 0; i < 64; ++i) mu += xr[i];
        mu *= (1.f / 64.f);
        float var = 0.f;
        for (int i = 0; i < 64; ++i) { float d = xr[i] - mu; var += d * d; }
        var *= (1.f / 64.f);
        const float rs = rsqrtf(var + 1e-5f);
#pragma unroll
        for (int i0 = 0; i0 < 64; i0 += 4) {
            float4 o;
            o.x = (xr[i0 + 0] - mu) * rs * tln_s[i0 + 0] + tln_b[i0 + 0];
            o.y = (xr[i0 + 1] - mu) * rs * tln_s[i0 + 1] + tln_b[i0 + 1];
            o.z = (xr[i0 + 2] - mu) * rs * tln_s[i0 + 2] + tln_b[i0 + 2];
            o.w = (xr[i0 + 3] - mu) * rs * tln_s[i0 + 3] + tln_b[i0 + 3];
            *(float4*)(tg + r * 64 + i0) = o;
        }
    }
}

// ---------------------------------------------------------------------------
// Kernel 3: slot attention + spatial top-2 mask + classification (tiny work)
// ---------------------------------------------------------------------------
#define K3_LDS 69632
__device__ __forceinline__ int sc_cell(int n) {
    const int r = n / 14, c = n % 14;
    const int rr = (r < 5) ? 0 : ((r < 10) ? 1 : 2);
    const int cc = (c < 5) ? 0 : ((c < 10) ? 1 : 2);
    return rr * 3 + cc;
}

__global__ void sc_slots(const float* __restrict__ tokens,
                         const float* __restrict__ slot_q,
                         const float* __restrict__ cls_w, const float* __restrict__ cls_b,
                         const int* __restrict__ use_spmask, const int* __restrict__ tau_p,
                         float* __restrict__ out) {
    extern __shared__ char smem[];
    float* TOK  = (float*)smem;             // 196 x 64
    float* SQN  = (float*)(smem + 50176);   // 12 x 64
    float* INV  = (float*)(smem + 53248);   // 196
    float* A    = (float*)(smem + 54272);   // 12 x 196
    float* MASS = (float*)(smem + 63744);   // 12 x 9
    int*   TOP  = (int*)  (smem + 64192);   // 12 x 2
    float* MR   = (float*)(smem + 64320);   // 12
    float* LSE  = (float*)(smem + 64384);   // 12
    float* SBUF = (float*)(smem + 64448);   // 12 x 64
    float* PV   = (float*)(smem + 67520);   // 12

    const int b = blockIdx.x, tid = threadIdx.x, nth = blockDim.x;
    const int usp = use_spmask[0];
    const float tau = fmaxf(1e-6f, (float)tau_p[0]);

    copy_v4((u32x4*)TOK, (const u32x4*)(tokens + (size_t)b * 196 * 64), 196 * 64 / 4, tid, nth);
    __syncthreads();

    for (int n = tid; n < 196; n += nth) {
        float s = 0.f;
        for (int d = 0; d < 64; ++d) { float v = TOK[n * 64 + d]; s += v * v; }
        INV[n] = 1.f / fmaxf(sqrtf(s), 1e-12f);
    }
    if (tid < 12) {
        float s = 0.f;
        for (int d = 0; d < 64; ++d) { float v = slot_q[tid * 64 + d]; s += v * v; }
        const float inv = 1.f / fmaxf(sqrtf(s), 1e-12f);
        for (int d = 0; d < 64; ++d) SQN[tid * 64 + d] = slot_q[tid * 64 + d] * inv;
    }
    __syncthreads();

    for (int i = tid; i < 12 * 196; i += nth) {
        const int m = i / 196, n = i % 196;
        float s = 0.f;
        for (int d = 0; d < 64; ++d) s += SQN[m * 64 + d] * TOK[n * 64 + d];
        A[i] = s * INV[n] * 0.125f;
    }
    __syncthreads();

    if (tid < 12) {
        float* r = A + tid * 196;
        float mx = -1e30f;
        for (int n = 0; n < 196; ++n) mx = fmaxf(mx, r[n]);
        float sm = 0.f;
        for (int n = 0; n < 196; ++n) { float e = expf(r[n] - mx); r[n] = e; sm += e; }
        LSE[tid] = mx + logf(sm);
        const float inv = 1.f / sm;
        for (int n = 0; n < 196; ++n) r[n] *= inv;
        float* ms = MASS + tid * 9;
        for (int k = 0; k < 9; ++k) ms[k] = 0.f;
        for (int n = 0; n < 196; ++n) ms[sc_cell(n)] += r[n];
        int i0 = 0;
        for (int k = 1; k < 9; ++k) if (ms[k] > ms[i0]) i0 = k;
        int i1 = (i0 == 0) ? 1 : 0;
        for (int k = 0; k < 9; ++k) if (k != i0 && ms[k] > ms[i1]) i1 = k;
        TOP[tid * 2] = i0; TOP[tid * 2 + 1] = i1;
        float mr = 0.f;
        for (int n = 0; n < 196; ++n) {
            const int cl = sc_cell(n);
            if (cl == i0 || cl == i1) mr += r[n];
        }
        MR[tid] = fmaxf(mr, 1e-8f);
    }
    __syncthreads();

    for (int i = tid; i < 12 * 64; i += nth) {
        const int m = i / 64, d = i % 64;
        const int c0 = TOP[m * 2], c1 = TOP[m * 2 + 1];
        const float inv = 1.f / MR[m];
        float s = 0.f;
        for (int n = 0; n < 196; ++n) {
            float w = A[m * 196 + n];
            if (usp) {
                const int cl = sc_cell(n);
                w = (cl == c0 || cl == c1) ? w * inv : 0.f;
            }
            s += w * TOK[n * 64 + d];
        }
        SBUF[i] = s;
    }
    __syncthreads();

    if (tid < 12) {
        float s = 0.f;
        for (int d = 0; d < 64; ++d) { float v = SBUF[tid * 64 + d]; s += v * v; }
        const float inv = 1.f / fmaxf(sqrtf(s), 1e-12f);
        for (int d = 0; d < 64; ++d) SBUF[tid * 64 + d] *= inv;
    }
    if (tid == 0) {
        float zs[12], zm = 0.f;
        for (int m = 0; m < 12; ++m) { zs[m] = usp ? MR[m] : LSE[m]; zm += zs[m]; }
        zm *= (1.f / 12.f);
        float mx = -1e30f;
        for (int m = 0; m < 12; ++m) { zs[m] = (zs[m] - zm) / tau; mx = fmaxf(mx, zs[m]); }
        float sm = 0.f;
        for (int m = 0; m < 12; ++m) { float e = expf(zs[m] - mx); PV[m] = e; sm += e; }
        for (int m = 0; m < 12; ++m) PV[m] /= sm;
    }
    __syncthreads();

    for (int c = tid; c < 47; c += nth) {
        float acc = 0.f;
        for (int m = 0; m < 12; ++m) {
            float s = 0.f;
            for (int d = 0; d < 64; ++d) s += SBUF[m * 64 + d] * cls_w[c * 64 + d];
            acc += (s + cls_b[c]) * PV[m];
        }
        out[b * 47 + c] = acc;
    }
}

// ---------------------------------------------------------------------------
// Launch
// ---------------------------------------------------------------------------
extern "C" void kernel_launch(void* const* d_in, const int* in_sizes, int n_in,
                              void* d_out, int out_size, void* d_ws, size_t ws_size,
                              hipStream_t stream) {
    (void)in_sizes; (void)n_in; (void)out_size; (void)ws_size;
    const float* x          = (const float*)d_in[0];
    const float* conv1_w    = (const float*)d_in[1];
    const float* proj_w     = (const float*)d_in[2];
    const float* proj_b     = (const float*)d_in[3];
    const float* ln1_s      = (const float*)d_in[4];
    const float* ln1_b      = (const float*)d_in[5];
    const float* attn_in_w  = (const float*)d_in[6];
    const float* attn_in_b  = (const float*)d_in[7];
    const float* attn_out_w = (const float*)d_in[8];
    const float* attn_out_b = (const float*)d_in[9];
    const float* ln2_s      = (const float*)d_in[10];
    const float* ln2_b      = (const float*)d_in[11];
    const float* ffn_w1     = (const float*)d_in[12];
    const float* ffn_b1     = (const float*)d_in[13];
    const float* ffn_w2     = (const float*)d_in[14];
    const float* ffn_b2     = (const float*)d_in[15];
    const float* tln_s      = (const float*)d_in[16];
    const float* tln_b      = (const float*)d_in[17];
    const float* slot_q     = (const float*)d_in[18];
    const float* cls_w      = (const float*)d_in[19];
    const float* cls_b      = (const float*)d_in[20];
    const int*   use_spmask = (const int*)d_in[21];
    const int*   tau_p      = (const int*)d_in[23];

    char* ws = (char*)d_ws;
    bf16_t* convB = (bf16_t*)(ws + WS_CONVB);
    bf16_t* projB = (bf16_t*)(ws + WS_PROJB);
    bf16_t* qkvB  = (bf16_t*)(ws + WS_QKVB);
    bf16_t* outB  = (bf16_t*)(ws + WS_OUTB);
    bf16_t* ff1B  = (bf16_t*)(ws + WS_FF1B);
    bf16_t* ff2B  = (bf16_t*)(ws + WS_FF2B);
    float*  tokw  = (float*)(ws + WS_TOKENS);

    sc_prep<<<(74752 + 255) / 256, 256, 0, stream>>>(
        conv1_w, proj_w, attn_in_w, attn_out_w, ffn_w1, ffn_w2,
        convB, projB, qkvB, outB, ff1B, ff2B);

    sc_conv_proj<<<512, 256, K1_LDS, stream>>>(x, convB, projB, proj_b, tokw);

    sc_transformer<<<512, 256, K2_LDS, stream>>>(
        tokw, qkvB, outB, ff1B, ff2B,
        ln1_s, ln1_b, attn_in_b, attn_out_b,
        ln2_s, ln2_b, ffn_b1, ffn_b2, tln_s, tln_b);

    sc_slots<<<512, 256, K3_LDS, stream>>>(
        tokw, slot_q, cls_w, cls_b, use_spmask, tau_p, (float*)d_out);
}
